// SAM2Mask_21191368638470
// MI455X (gfx1250) — matrix operand order repaired and checked by reading, not compiled, not run on hardware
//
#include <hip/hip_runtime.h>
#include <hip/hip_bf16.h>
#include <math.h>

#define S_SP 512
#define P_PTS 512
#define K_FPS 10
#define NUM_MASKS 128
#define H_IMG 512
#define W_IMG 512
#define FEAT_DIM 32
#define RADIUS 0.2f
#define MASK_THRES 0.2f
#define SIM_THRES 0.5f
#define N_MASK 5
#define TOP_K 10

typedef __attribute__((ext_vector_type(16))) _Float16 v16h;
typedef __attribute__((ext_vector_type(8)))  float    v8f;

// ---------------------------------------------------------------------------
// Kernel 1: farthest-point sampling, 10 reps per superpoint (block = superpoint)
// ---------------------------------------------------------------------------
__global__ __launch_bounds__(256) void fps_kernel(const float* __restrict__ pcds,
                                                  float* __restrict__ reps) {
    __shared__ float sx[P_PTS], sy[P_PTS], sz[P_PTS], sd[P_PTS];
    __shared__ float rv[256];
    __shared__ int   ri[256];
    const int s = blockIdx.x;
    const int t = threadIdx.x;
    const float* base = pcds + (size_t)s * P_PTS * 3;

    for (int p = t; p < P_PTS; p += 256) {
        sx[p] = base[p * 3 + 0];
        sy[p] = base[p * 3 + 1];
        sz[p] = base[p * 3 + 2];
    }
    __syncthreads();

    const float x0 = sx[0], y0 = sy[0], z0 = sz[0];
    for (int p = t; p < P_PTS; p += 256) {
        float dx = sx[p] - x0, dy = sy[p] - y0, dz = sz[p] - z0;
        sd[p] = dx * dx + dy * dy + dz * dz;   // squared dist: same argmax/min order
    }
    if (t == 0) {
        reps[(s * K_FPS + 0) * 3 + 0] = x0;
        reps[(s * K_FPS + 0) * 3 + 1] = y0;
        reps[(s * K_FPS + 0) * 3 + 2] = z0;
    }
    __syncthreads();

    for (int it = 1; it < K_FPS; ++it) {
        // per-thread first-max over strided slice
        float bv = -1.0f; int bi = P_PTS;
        for (int p = t; p < P_PTS; p += 256) {
            float v = sd[p];
            if (v > bv) { bv = v; bi = p; }
        }
        rv[t] = bv; ri[t] = bi;
        __syncthreads();
        // tree reduce, tie -> smaller index (matches jnp.argmax first-occurrence)
        for (int off = 128; off > 0; off >>= 1) {
            if (t < off) {
                float ov = rv[t + off]; int oi = ri[t + off];
                if (ov > rv[t] || (ov == rv[t] && oi < ri[t])) { rv[t] = ov; ri[t] = oi; }
            }
            __syncthreads();
        }
        const int idx = ri[0];
        const float xi = sx[idx], yi = sy[idx], zi = sz[idx];
        if (t == 0) {
            reps[(s * K_FPS + it) * 3 + 0] = xi;
            reps[(s * K_FPS + it) * 3 + 1] = yi;
            reps[(s * K_FPS + it) * 3 + 2] = zi;
        }
        for (int p = t; p < P_PTS; p += 256) {
            float dx = sx[p] - xi, dy = sy[p] - yi, dz = sz[p] - zi;
            float d = dx * dx + dy * dy + dz * dz;
            sd[p] = fminf(sd[p], d);
        }
        __syncthreads();
    }
}

// ---------------------------------------------------------------------------
// Kernel 2: base adjacency from min distance between 10-rep sets
// ---------------------------------------------------------------------------
__global__ __launch_bounds__(256) void adjbase_kernel(const float* __restrict__ reps,
                                                      float* __restrict__ adj) {
    const int gid = blockIdx.x * 256 + threadIdx.x;   // S*S threads
    const int i = gid >> 9;
    const int j = gid & (S_SP - 1);
    const float* ra = reps + (size_t)i * K_FPS * 3;
    const float* rb = reps + (size_t)j * K_FPS * 3;
    float ax[K_FPS], ay[K_FPS], az[K_FPS];
    float bx[K_FPS], by[K_FPS], bz[K_FPS];
#pragma unroll
    for (int a = 0; a < K_FPS; ++a) {
        ax[a] = ra[a * 3 + 0]; ay[a] = ra[a * 3 + 1]; az[a] = ra[a * 3 + 2];
        bx[a] = rb[a * 3 + 0]; by[a] = rb[a * 3 + 1]; bz[a] = rb[a * 3 + 2];
    }
    float d2min = 3.4e38f;
#pragma unroll
    for (int a = 0; a < K_FPS; ++a) {
#pragma unroll
        for (int b = 0; b < K_FPS; ++b) {
            float dx = ax[a] - bx[b], dy = ay[a] - by[b], dz = az[a] - bz[b];
            float d2 = dx * dx + dy * dy + dz * dz;
            d2min = fminf(d2min, d2);
        }
    }
    const float dmin = sqrtf(fmaxf(d2min, 1e-12f));
    float v = 0.0f;
    if (dmin < RADIUS && i != j) v = expf(-dmin);
    adj[gid] = v;
}

// ---------------------------------------------------------------------------
// Kernel 3a: bitplane transpose of masks: pixel-major 128-bit rows (4 x u32).
// Coalesced single pass over the 128 MB mask array -> 4 MB packed buffer.
// ---------------------------------------------------------------------------
__global__ __launch_bounds__(256) void pack_masks_kernel(const int* __restrict__ masks,
                                                         unsigned* __restrict__ packed) {
    const int pix = blockIdx.x * 256 + threadIdx.x;   // H*W threads
    unsigned wds[4] = {0u, 0u, 0u, 0u};
#pragma unroll 4
    for (int m = 0; m < NUM_MASKS; ++m) {
        const unsigned b = (masks[(size_t)m * H_IMG * W_IMG + pix] != 0) ? 1u : 0u;
        wds[m >> 5] |= b << (m & 31);
    }
    uint4* dst = (uint4*)packed;
    dst[pix] = make_uint4(wds[0], wds[1], wds[2], wds[3]);
}

// ---------------------------------------------------------------------------
// Kernel 3b: per-superpoint per-mask counts from packed bitplanes.
// Block = superpoint, thread = mask. All 32 lanes of a wave probe the SAME
// packed word per point -> single broadcast L2 transaction per wave-iter.
// ---------------------------------------------------------------------------
__global__ __launch_bounds__(128) void counts_packed_kernel(const unsigned* __restrict__ packed,
                                                            const int* __restrict__ mapping,
                                                            int* __restrict__ counts) {
    __shared__ int spix[P_PTS];
    __shared__ int svis[P_PTS];
    const int s = blockIdx.x;
    const int t = threadIdx.x;   // mask id
    for (int p = t; p < P_PTS; p += 128) {
        const int base = (s * P_PTS + p) * 4;
        int y = mapping[base + 1]; y = min(max(y, 0), H_IMG - 1);
        int x = mapping[base + 2]; x = min(max(x, 0), W_IMG - 1);
        spix[p] = y * W_IMG + x;
        svis[p] = (mapping[base + 3] == 1) ? 1 : 0;
    }
    __syncthreads();
    const int w = t >> 5;                 // uniform within a wave
    const unsigned bit = 1u << (t & 31);
    int cnt = 0;
    for (int p = 0; p < P_PTS; ++p) {
        if (svis[p]) cnt += (packed[(size_t)spix[p] * 4 + w] & bit) ? 1 : 0;
    }
    counts[s * NUM_MASKS + t] = cnt;
}

// ---------------------------------------------------------------------------
// Kernel 3 (fallback, small ws): direct strided mask probing (L2-resident)
// ---------------------------------------------------------------------------
__global__ __launch_bounds__(128) void counts_kernel(const int* __restrict__ masks,
                                                     const int* __restrict__ mapping,
                                                     int* __restrict__ counts) {
    __shared__ int spix[P_PTS];
    __shared__ int svis[P_PTS];
    const int s = blockIdx.x;
    const int t = threadIdx.x;   // mask id
    for (int p = t; p < P_PTS; p += 128) {
        const int base = (s * P_PTS + p) * 4;
        int y = mapping[base + 1]; y = min(max(y, 0), H_IMG - 1);
        int x = mapping[base + 2]; x = min(max(x, 0), W_IMG - 1);
        spix[p] = y * W_IMG + x;
        svis[p] = (mapping[base + 3] == 1) ? 1 : 0;
    }
    __syncthreads();
    const int* mbase = masks + (size_t)t * H_IMG * W_IMG;
    int cnt = 0;
    for (int p = 0; p < P_PTS; ++p) {
        if (svis[p]) cnt += (mbase[spix[p]] != 0) ? 1 : 0;
    }
    counts[s * NUM_MASKS + t] = cnt;
}

// ---------------------------------------------------------------------------
// Kernel 4: stable top-10 superpoints per mask -> keep bitmask (thread = mask)
// ---------------------------------------------------------------------------
__global__ __launch_bounds__(128) void topk_kernel(const int* __restrict__ counts,
                                                   unsigned* __restrict__ keep) {
    const int m = threadIdx.x;
    unsigned kw[16];
#pragma unroll
    for (int w = 0; w < 16; ++w) kw[w] = 0u;
    for (int r = 0; r < TOP_K; ++r) {
        int bestv = -1, bests = 0;
        for (int s = 0; s < S_SP; ++s) {
            if ((kw[s >> 5] >> (s & 31)) & 1u) continue;
            int v = counts[s * NUM_MASKS + m];
            if (v > bestv) { bestv = v; bests = s; }   // strict > : stable tie-break
        }
        kw[bests >> 5] |= 1u << (bests & 31);
    }
#pragma unroll
    for (int w = 0; w < 16; ++w) keep[m * 16 + w] = kw[w];
}

// ---------------------------------------------------------------------------
// Kernel 5: frac_kept -> contrib bitmask (128 bits per superpoint) + conf
// ---------------------------------------------------------------------------
__global__ __launch_bounds__(256) void contrib_conf_kernel(const int* __restrict__ counts,
                                                           const unsigned* __restrict__ keep,
                                                           const float* __restrict__ sam,
                                                           unsigned* __restrict__ contrib,
                                                           float* __restrict__ conf) {
    const int s = blockIdx.x * 256 + threadIdx.x;
    if (s >= S_SP) return;
    unsigned c[4] = {0u, 0u, 0u, 0u};
    float sumw = 0.0f, sumf = 0.0f;
    const float denom = (float)P_PTS + 1e-6f;   // == 512.0f in f32, same as ref
    for (int m = 0; m < NUM_MASKS; ++m) {
        const unsigned kept = (keep[m * 16 + (s >> 5)] >> (s & 31)) & 1u;
        if (kept) {
            const float f = (float)counts[s * NUM_MASKS + m] / denom;
            sumf += f;
            sumw += f * sam[m];
            if (f >= MASK_THRES) c[m >> 5] |= 1u << (m & 31);
        }
    }
#pragma unroll
    for (int w = 0; w < 4; ++w) contrib[s * 4 + w] = c[w];
    conf[s] = fmaxf(sumw / (sumf + 1e-6f), 0.0f);
}

// ---------------------------------------------------------------------------
// Kernel 6: superpoint features: mean over 512 points, L2-normalize, -> f16
// (block = superpoint, 32 threads = 32 dims, wave shuffle reduction)
// ---------------------------------------------------------------------------
__global__ __launch_bounds__(32) void feats_kernel(const float* __restrict__ pf,
                                                   _Float16* __restrict__ fh) {
    const int s = blockIdx.x;
    const int d = threadIdx.x;
    const float* base = pf + (size_t)s * P_PTS * FEAT_DIM + d;
    float acc = 0.0f;
    for (int p = 0; p < P_PTS; ++p) acc += base[(size_t)p * FEAT_DIM];
    const float mean = acc / ((float)P_PTS + 1e-6f);
    float sq = mean * mean;
#pragma unroll
    for (int off = 16; off > 0; off >>= 1) sq += __shfl_xor(sq, off, 32);
    const float nrm = fmaxf(sqrtf(sq), 1e-12f);
    fh[s * FEAT_DIM + d] = (_Float16)(mean / nrm);
}

// ---------------------------------------------------------------------------
// Kernel 7: sim = feats @ feats^T via v_wmma_f32_16x16x32_f16 (K=32 exact),
// fused with top-5-of-union IoU and adjacency modulation.
// One wave per 16x16 tile; 8 waves/block; 1024 tiles total.
// ---------------------------------------------------------------------------
__global__ __launch_bounds__(256) void sim_mod_kernel(const _Float16* __restrict__ fh,
                                                      const unsigned* __restrict__ contrib,
                                                      float* __restrict__ adj) {
    const int wave = threadIdx.x >> 5;
    const int lane = threadIdx.x & 31;
    const int gw = blockIdx.x * 8 + wave;
    const int tm = gw >> 5;      // 32 tile-rows
    const int tn = gw & 31;      // 32 tile-cols
    const int half = lane >> 4;  // 0: lanes 0-15, 1: lanes 16-31
    const int l15 = lane & 15;

    const _Float16* ra = fh + (size_t)(tm * 16 + l15) * FEAT_DIM;
    const _Float16* rb = fh + (size_t)(tn * 16 + l15) * FEAT_DIM;

    // A 16x32 f16 layout: lanes 0-15 hold K[0..7],K[16..23]; lanes 16-31 K[8..15],K[24..31]
    v16h a, b;
    const int ka = half * 8;
#pragma unroll
    for (int q = 0; q < 8; ++q) {
        a[q]     = ra[ka + q];
        a[8 + q] = ra[16 + ka + q];
    }
    // B 32x16 f16 layout: lanes 0-15 hold K[0..7],K[16..23]; lanes 16-31 K[8..15],K[24..31]
    const int kb = half * 8;
#pragma unroll
    for (int q = 0; q < 8; ++q) {
        b[q]     = rb[kb + q];
        b[8 + q] = rb[16 + kb + q];
    }

    v8f c = {};
    c = __builtin_amdgcn_wmma_f32_16x16x32_f16(false, a, false, b, (short)0, c, false, false);

    const int N = tn * 16 + l15;
    const unsigned cj0 = contrib[N * 4 + 0];
    const unsigned cj1 = contrib[N * 4 + 1];
    const unsigned cj2 = contrib[N * 4 + 2];
    const unsigned cj3 = contrib[N * 4 + 3];

#pragma unroll
    for (int v = 0; v < 8; ++v) {
        const int M = tm * 16 + v + half * 8;   // C layout: VGPR v -> M=v (lanes<16) / v+8
        float simv = (c[v] + 1.0f) * 0.5f;
        simv = fminf(fmaxf(simv, 0.0f), 1.0f);
        if (simv < SIM_THRES) simv = 1.0f;

        unsigned un[4], iw[4];
        const unsigned ci0 = contrib[M * 4 + 0];
        const unsigned ci1 = contrib[M * 4 + 1];
        const unsigned ci2 = contrib[M * 4 + 2];
        const unsigned ci3 = contrib[M * 4 + 3];
        un[0] = ci0 | cj0; un[1] = ci1 | cj1; un[2] = ci2 | cj2; un[3] = ci3 | cj3;
        iw[0] = ci0 & cj0; iw[1] = ci1 & cj1; iw[2] = ci2 & cj2; iw[3] = ci3 & cj3;

        const int u = __popc(un[0]) + __popc(un[1]) + __popc(un[2]) + __popc(un[3]);
        const int t5 = (u < N_MASK) ? u : N_MASK;    // uni over selected indices
        // stable top-5 of union == first t5 set bits in ascending mask index
        int icnt = 0, taken = 0;
#pragma unroll
        for (int w = 0; w < 4; ++w) {
            unsigned uw = un[w];
            const unsigned ww = iw[w];
            while (uw && taken < N_MASK) {
                const unsigned lsb = uw & (unsigned)(-(int)uw);
                if (ww & lsb) ++icnt;
                uw ^= lsb;
                ++taken;
            }
        }
        const float overlap = (float)icnt / ((float)t5 + 1e-6f);
        const float mod = overlap * simv;

        const size_t off = (size_t)M * S_SP + N;
        const float a0 = adj[off];
        adj[off] = (a0 > 0.0f) ? fmaxf(mod, a0) : a0;
    }
}

// ---------------------------------------------------------------------------
extern "C" void kernel_launch(void* const* d_in, const int* in_sizes, int n_in,
                              void* d_out, int out_size, void* d_ws, size_t ws_size,
                              hipStream_t stream) {
    (void)in_sizes; (void)n_in; (void)out_size;

    const float* pcds    = (const float*)d_in[0];
    // d_in[1] = spt_labels (implicit: sorted contiguous, unused)
    const int*   masks   = (const int*)d_in[2];
    const int*   mapping = (const int*)d_in[3];
    const float* sam     = (const float*)d_in[4];
    const float* pfeas   = (const float*)d_in[5];

    float* out  = (float*)d_out;           // adjacency (S*S) then conf (S)
    float* conf = out + (size_t)S_SP * S_SP;

    char* ws = (char*)d_ws;
    float*     reps    = (float*)(ws + 0);          // 512*10*3 f32   = 61440 B
    int*       counts  = (int*)(ws + 61440);        // 512*128 i32    = 262144 B
    unsigned*  keep    = (unsigned*)(ws + 323584);  // 128*16 u32     = 8192 B
    unsigned*  contrib = (unsigned*)(ws + 331776);  // 512*4 u32      = 8192 B
    _Float16*  fh      = (_Float16*)(ws + 339968);  // 512*32 f16     = 32768 B
    unsigned*  packed  = (unsigned*)(ws + 372736);  // H*W*4 u32      = 4194304 B
    const size_t ws_needed_packed = 372736u + (size_t)H_IMG * W_IMG * 4 * sizeof(unsigned);

    fps_kernel<<<S_SP, 256, 0, stream>>>(pcds, reps);
    adjbase_kernel<<<(S_SP * S_SP) / 256, 256, 0, stream>>>(reps, out);

    if (ws_size >= ws_needed_packed) {
        // preferred path: bitplane transpose (coalesced 128 MB pass) + broadcast probes
        pack_masks_kernel<<<(H_IMG * W_IMG) / 256, 256, 0, stream>>>(masks, packed);
        counts_packed_kernel<<<S_SP, 128, 0, stream>>>(packed, mapping, counts);
    } else {
        counts_kernel<<<S_SP, 128, 0, stream>>>(masks, mapping, counts);
    }

    topk_kernel<<<1, 128, 0, stream>>>(counts, keep);
    contrib_conf_kernel<<<2, 256, 0, stream>>>(counts, keep, sam, contrib, conf);
    feats_kernel<<<S_SP, 32, 0, stream>>>(pfeas, fh);
    sim_mod_kernel<<<128, 256, 0, stream>>>(fh, contrib, out);
}
